// M_SSM_1726576853497
// MI455X (gfx1250) — compile-verified, matrix operand order checked by reading, not run on hardware
//
#include <hip/hip_runtime.h>
#include <hip/hip_bf16.h>
#include <math.h>

#define L4 4096        // H*W
#define NB 4           // batch

typedef __attribute__((ext_vector_type(16))) __bf16    v16bf;
typedef __attribute__((ext_vector_type(8)))  float     v8f;
typedef __attribute__((ext_vector_type(4)))  unsigned  v4u;
typedef __attribute__((ext_vector_type(8)))  int       v8i;
typedef __attribute__((ext_vector_type(4)))  int       v4i;

#if defined(__HIP_DEVICE_COMPILE__) && __has_builtin(__builtin_amdgcn_tensor_load_to_lds) && __has_builtin(__builtin_amdgcn_s_wait_tensorcnt)
#define USE_TDM 1
#else
#define USE_TDM 0
#endif

__device__ __forceinline__ float sigf(float x){ return 1.f/(1.f+expf(-x)); }
__device__ __forceinline__ float softplusf(float x){ return x > 20.f ? x : log1pf(expf(x)); }

// map scan position l -> index into xc ([B,128,H*W], hw order) for direction k
__device__ __forceinline__ int dir_map(int k, int l){
  if (k & 2) l = (L4-1) - l;
  if (k & 1) l = ((l & 63) << 6) | (l >> 6);
  return l;
}

#if USE_TDM
// TDM: DMA a 16-row x tileCols-col fp32 tile (row stride rowStride) into LDS.
// Rows beyond remRows / cols beyond remCols read as zero (hardware OOB rule).
__device__ __forceinline__ void tdm_load_b(const float* gsrc, unsigned lds_byte,
                                           int remCols, int remRows, int rowStride,
                                           int tileCols)
{
  unsigned long long ga = (unsigned long long)(size_t)gsrc;
  unsigned td0 = (unsigned)remCols, td1 = (unsigned)(remRows > 0 ? remRows : 0);
  v4u g0; v8i g1; v4i g2 = {0,0,0,0}; v4i g3 = {0,0,0,0};
  v8i g4 = {0,0,0,0,0,0,0,0};
  g0[0] = 1u;                                                  // count=1, user mode
  g0[1] = lds_byte;                                            // lds_addr
  g0[2] = (unsigned)(ga & 0xFFFFFFFFu);                        // global_addr[31:0]
  g0[3] = (unsigned)((ga >> 32) & 0x01FFFFFFu) | (2u << 30);   // addr[56:32] | type=2
  g1[0] = 0x20000;                                             // data_size=4B, no flags
  g1[1] = (int)((td0 & 0xFFFFu) << 16);                        // tensor_dim0 lo
  g1[2] = (int)((td0 >> 16) | ((td1 & 0xFFFFu) << 16));        // dim0 hi | dim1 lo
  g1[3] = (int)((td1 >> 16) | ((unsigned)tileCols << 16));     // dim1 hi | tile_dim0
  g1[4] = 16;                                                  // tile_dim1=16, tile_dim2=0
  g1[5] = (int)(unsigned)rowStride;                            // tensor_dim0_stride lo32
  g1[6] = 0;
  g1[7] = 0;
  __builtin_amdgcn_tensor_load_to_lds(g0, g1, g2, g3, g4, 0);
}
#endif

// ---------------------------------------------------------------------------
// WMMA GEMM, 256 threads = 8 waves; each wave one 16x16 C tile (8 M-tiles/block,
// shared N-tile). B panel (16 x up-to-128 k) staged in LDS via TDM (or coop
// loads), zero-padded for OOB rows. C[row,col] = sum_k A(row,k)*Bt[col][k].
//   CONV: implicit-GEMM 3x3 SAME conv with TAP-MAJOR k: k = tap*128 + ci
//         (B must be pre-packed as [co][tap][ci]).
//   epi: 0 store, 1 y=relu(y*sc[col]+sh[col]), 2 y += res[...]
// ---------------------------------------------------------------------------
template<bool CONV>
__global__ __launch_bounds__(256)
void wmma_gemm_k(const float* __restrict__ A, const float* __restrict__ Bt,
                 float* __restrict__ C,
                 int M, int N, int K,
                 int permMode, int As_row, int As_col, long long batchA,
                 int Bt_n, long long batchB,
                 int Cs_r, int Cs_c, long long batchC,
                 const float* __restrict__ sc, const float* __restrict__ sh,
                 const float* __restrict__ res, long long batchRes, int epi)
{
  __shared__ float ldsB[16*128];
  const int lane = threadIdx.x & 31;
  const int wv   = threadIdx.x >> 5;          // 0..7
  const int half = lane >> 4;                 // K-block select in 16-bit A/B layout
  const int lm   = lane & 15;
  const int tM = blockIdx.x*8 + wv, tN = blockIdx.y, bz = blockIdx.z;
  const float* Ab = A + (long long)bz * batchA;
  const float* Bb = Bt + (long long)bz * batchB;
  float* Cb = C + (long long)bz * batchC;

  const int rowA = tM*16 + lm;                // M is a multiple of 128 here
  long long rowoff = 0;
  int convH = 0, convW = 0;
  if (CONV) { convH = rowA >> 6; convW = rowA & 63; }
  else {
    int pr = rowA;
    if      (permMode == 1) pr = ((rowA & 63) << 6) | (rowA >> 6);
    else if (permMode == 2) pr = (L4-1) - rowA;
    else if (permMode == 3) { int lp = (L4-1) - rowA; pr = ((lp & 63) << 6) | (lp >> 6); }
    rowoff = (long long)pr * As_row;
  }
  const int colC = tN*16 + lm;

  v8f acc = {};
  for (int kp = 0; kp < K; kp += 128) {
    int pk = K - kp; if (pk > 128) pk = 128;

    __syncthreads();                          // previous panel fully consumed
#if USE_TDM
    if (wv == 0) {
      tdm_load_b(Bb + (long long)tN*16*Bt_n + kp, (unsigned)(size_t)ldsB,
                 Bt_n - kp, N - tN*16, Bt_n, pk);
      __builtin_amdgcn_s_wait_tensorcnt(0);
    }
#else
    for (int i = threadIdx.x; i < 16*pk; i += 256) {
      int r = i / pk, cc = i - r*pk;
      int gr = tN*16 + r;
      ldsB[i] = (gr < N) ? Bb[(long long)gr*Bt_n + kp + cc] : 0.f;
    }
#endif
    __syncthreads();

    for (int kk = kp; kk < kp+pk; kk += 32) {
      v16bf af, bfr;
      // ---- A fragment (global) ----
      if (CONV) {
        const int tap = kk >> 7;              // constant within this 32-K step
        const int t3  = tap/3;
        const int hh  = convH + t3 - 1;
        const int ww  = convW + (tap - t3*3) - 1;
        const bool inb = ((unsigned)hh < 64u) & ((unsigned)ww < 64u);
        const float* srcA = Ab + ((hh<<6) + ww);
        const int cb = (kk & 127) + half*8;   // ci base for this fragment
        float tA[16];
        if (inb) {
          #pragma unroll
          for (int j = 0; j < 16; ++j)
            tA[j] = srcA[(long long)(cb + ((j<8)?j:(8+j)))*L4];
        } else {
          #pragma unroll
          for (int j = 0; j < 16; ++j) tA[j] = 0.f;
        }
        #pragma unroll
        for (int j = 0; j < 16; ++j) af[j] = (__bf16)tA[j];
      } else {
        const int kb = kk + half*8;
        #pragma unroll
        for (int j = 0; j < 16; ++j) {
          int k = kb + ((j < 8) ? j : (8 + j));   // K: kb+0..7, kb+16..23
          af[j] = (__bf16)Ab[rowoff + (long long)k*As_col];
        }
      }
      // ---- B fragment (LDS, vectorized; zero-padded rows) ----
      {
        const float* brow = &ldsB[lm*pk + (kk - kp) + half*8];
        float4 b0 = *(const float4*)(brow);
        float4 b1 = *(const float4*)(brow + 4);
        float4 b2 = *(const float4*)(brow + 16);
        float4 b3 = *(const float4*)(brow + 20);
        bfr[0]=(__bf16)b0.x; bfr[1]=(__bf16)b0.y; bfr[2]=(__bf16)b0.z; bfr[3]=(__bf16)b0.w;
        bfr[4]=(__bf16)b1.x; bfr[5]=(__bf16)b1.y; bfr[6]=(__bf16)b1.z; bfr[7]=(__bf16)b1.w;
        bfr[8]=(__bf16)b2.x; bfr[9]=(__bf16)b2.y; bfr[10]=(__bf16)b2.z; bfr[11]=(__bf16)b2.w;
        bfr[12]=(__bf16)b3.x; bfr[13]=(__bf16)b3.y; bfr[14]=(__bf16)b3.z; bfr[15]=(__bf16)b3.w;
      }
      acc = __builtin_amdgcn_wmma_f32_16x16x32_bf16(false, af, false, bfr,
                                                    (short)0, acc, false, false);
    }
  }

  #pragma unroll
  for (int i = 0; i < 8; ++i) {
    int row = tM*16 + half*8 + i;             // C/D layout: vgpr i -> M=i / M=i+8
    if (colC < N) {
      long long cidx = (long long)row * Cs_r + (long long)colC * Cs_c;
      float v = acc[i];
      if      (epi == 1) v = fmaxf(v * sc[colC] + sh[colC], 0.f);
      else if (epi == 2) v += res[(long long)bz * batchRes + cidx];
      Cb[cidx] = v;
    }
  }
}

// ---------------------------------------------------------------------------
__global__ void pack_cat_k(const float* __restrict__ rgb, const float* __restrict__ t,
                           float* __restrict__ cat)
{
  int idx = blockIdx.x*blockDim.x + threadIdx.x;
  const int per = NB*64*L4;
  if (idx >= 2*per) return;
  const float* src = (idx < per) ? rgb : t;
  int off = (idx < per) ? idx : idx - per;
  int l = off & (L4-1);
  int c = (off >> 12) & 63;
  int b = off >> 18;
  int chan = ((idx < per) ? 64 : 128) + c;
  cat[((long long)b*192 + chan)*L4 + l] = src[off];
}

// repack c1_w [co][ci][tap] -> tap-major [co][tap][ci]  (tap = ky*3+kx)
__global__ void repack_c1w_k(const float* __restrict__ w, float* __restrict__ wp)
{
  int idx = blockIdx.x*blockDim.x + threadIdx.x;
  if (idx >= 64*1152) return;
  int co  = idx / 1152;
  int r   = idx - co*1152;
  int tap = r >> 7;          // 0..8
  int ci  = r & 127;
  wp[idx] = w[co*1152 + ci*9 + tap];
}

// fused BN coefficients: y = (acc + conv_b - m)*g/sqrt(v+eps) + beta = acc*sc + sh
__global__ void prep_bn_k(const float* c1b, const float* g1, const float* b1,
                          const float* m1, const float* v1,
                          const float* c2b, const float* g2, const float* b2,
                          const float* m2, const float* v2,
                          float* sc1, float* sh1, float* sc2, float* sh2)
{
  int i = threadIdx.x;
  if (i < 64) {
    float s = g1[i] * rsqrtf(v1[i] + 1e-5f);
    sc1[i] = s; sh1[i] = b1[i] + (c1b[i] - m1[i]) * s;
  } else if (i < 128) {
    int j = i - 64;
    float s = g2[j] * rsqrtf(v2[j] + 1e-5f);
    sc2[j] = s; sh2[j] = b2[j] + (c2b[j] - m2[j]) * s;
  }
}

// depthwise 3x3 SAME conv + bias + SiLU on xz channels 0..127 -> xc [B,128,L]
__global__ void dwconv_silu_k(const float* __restrict__ xz, const float* __restrict__ w,
                              const float* __restrict__ bias, float* __restrict__ xc,
                              int total)
{
  int idx = blockIdx.x*blockDim.x + threadIdx.x;
  if (idx >= total) return;
  int l  = idx & (L4-1);
  int t  = idx >> 12;
  int ci = t & 127;
  int b  = t >> 7;
  int h = l >> 6, wc = l & 63;
  float s = bias[ci];
  const float* src = xz + ((long long)b*256 + ci) * L4;
  #pragma unroll
  for (int ky = 0; ky < 3; ++ky) {
    int hh = h + ky - 1;
    if ((unsigned)hh >= 64u) continue;
    #pragma unroll
    for (int kx = 0; kx < 3; ++kx) {
      int ww = wc + kx - 1;
      if ((unsigned)ww >= 64u) continue;
      s += src[(hh<<6)+ww] * w[ci*9 + ky*3 + kx];
    }
  }
  xc[((long long)b*128 + ci)*L4 + l] = s * sigf(s);
}

// ---------------------------------------------------------------------------
// selective scan: 2048 scans (b,kdir,d) x 16 state lanes; lane n owns state n.
// ---------------------------------------------------------------------------
__global__ __launch_bounds__(256)
void scan_k(const float* __restrict__ xc, const float* __restrict__ xdbl,
            float* __restrict__ ys,
            const float* __restrict__ A_log, const float* __restrict__ dt_w,
            const float* __restrict__ dt_b, const float* __restrict__ Ds)
{
  int tid = blockIdx.x*blockDim.x + threadIdx.x;   // 0..32767
  int n  = tid & 15;
  int g  = tid >> 4;          // (b*4+k)*128 + d
  int d  = g & 127;
  int bk = g >> 7;
  int k  = bk & 3;
  int b  = bk >> 2;

  int kd = k*128 + d;
  float An  = -expf(A_log[kd*16 + n]);
  float Dv  = Ds[kd];
  float dtb = dt_b[kd];
  float w0 = dt_w[kd*4+0], w1 = dt_w[kd*4+1], w2 = dt_w[kd*4+2], w3 = dt_w[kd*4+3];

  const float* xcb = xc + ((long long)b*128 + d) * L4;
  const float* xd  = xdbl + (long long)bk * 36 * L4;
  float* yo = ys + ((long long)bk*128 + d) * L4;

  float hst = 0.f;
  for (int l = 0; l < L4; ++l) {
    __builtin_prefetch(&xd[(4+n)*L4  + l + 16], 0, 1);
    __builtin_prefetch(&xd[(20+n)*L4 + l + 16], 0, 1);
    int li  = dir_map(k, l);
    float u = xcb[li];
    float dt = dtb + xd[0*L4+l]*w0 + xd[1*L4+l]*w1 + xd[2*L4+l]*w2 + xd[3*L4+l]*w3;
    dt = softplusf(dt);
    float Bv = xd[(4+n)*L4 + l];
    float Cv = xd[(20+n)*L4 + l];
    hst = expf(dt*An)*hst + dt*u*Bv;
    float y = hst*Cv;
    y += __shfl_xor(y, 8);
    y += __shfl_xor(y, 4);
    y += __shfl_xor(y, 2);
    y += __shfl_xor(y, 1);
    if (n == 0) yo[l] = y + Dv*u;
  }
}

// ---------------------------------------------------------------------------
// combine 4 directions + LayerNorm(d=128) + SiLU gate; wave per (b,l)
// ---------------------------------------------------------------------------
__global__ __launch_bounds__(256)
void combine_ln_gate_k(const float* __restrict__ ys, const float* __restrict__ xz,
                       const float* __restrict__ ln_g, const float* __restrict__ ln_b,
                       float* __restrict__ yg)
{
  int wid  = blockIdx.x*8 + (threadIdx.x >> 5);  // b*L + l
  int lane = threadIdx.x & 31;
  int l = wid & (L4-1);
  int b = wid >> 12;
  int h = l >> 6, w = l & 63;
  int lt = (w << 6) | h;
  const float* y0 = ys + ((long long)(b*4+0)*128) * L4;
  const float* y1 = ys + ((long long)(b*4+1)*128) * L4;
  const float* y2 = ys + ((long long)(b*4+2)*128) * L4;
  const float* y3 = ys + ((long long)(b*4+3)*128) * L4;

  float v[4];
  float s = 0.f, s2 = 0.f;
  #pragma unroll
  for (int j = 0; j < 4; ++j) {
    int d = lane + 32*j;
    float tv = y0[(long long)d*L4 + l]
             + y1[(long long)d*L4 + lt]
             + y2[(long long)d*L4 + (L4-1-l)]
             + y3[(long long)d*L4 + (L4-1-lt)];
    v[j] = tv; s += tv; s2 += tv*tv;
  }
  #pragma unroll
  for (int m = 16; m >= 1; m >>= 1) { s += __shfl_xor(s, m); s2 += __shfl_xor(s2, m); }
  float mu   = s  * (1.f/128.f);
  float var  = s2 * (1.f/128.f) - mu*mu;
  float rstd = rsqrtf(var + 1e-5f);

  float* out = yg + (long long)wid * 128;
  const float* zb = xz + ((long long)b*256 + 128) * L4;   // z = xz channels 128..255
  #pragma unroll
  for (int j = 0; j < 4; ++j) {
    int d = lane + 32*j;
    float z   = zb[(long long)d*L4 + l];
    float val = (v[j]-mu)*rstd*ln_g[d] + ln_b[d];
    out[d] = val * (z * sigf(z));
  }
}

// ---------------------------------------------------------------------------
extern "C" void kernel_launch(void* const* d_in, const int* in_sizes, int n_in,
                              void* d_out, int out_size, void* d_ws, size_t ws_size,
                              hipStream_t stream)
{
  (void)in_sizes; (void)n_in; (void)out_size; (void)ws_size;
  const float* rgb  = (const float*)d_in[0];
  const float* tin  = (const float*)d_in[1];
  const float* c1w  = (const float*)d_in[2];
  const float* c1b  = (const float*)d_in[3];
  const float* bn1g = (const float*)d_in[4];
  const float* bn1b = (const float*)d_in[5];
  const float* bn1m = (const float*)d_in[6];
  const float* bn1v = (const float*)d_in[7];
  const float* c2w  = (const float*)d_in[8];
  const float* c2b  = (const float*)d_in[9];
  const float* bn2g = (const float*)d_in[10];
  const float* bn2b = (const float*)d_in[11];
  const float* bn2m = (const float*)d_in[12];
  const float* bn2v = (const float*)d_in[13];

  float* out = (float*)d_out;                 // [out | rgb_ | t_], each B*64*L4
  float* ws  = (float*)d_ws;

  float* sc1  = ws;       float* sh1 = ws + 64;
  float* sc2  = ws + 128; float* sh2 = ws + 192;
  float* c1wp = ws + 256;                     // 73,728 floats, tap-major conv1 weights
  float* arena = ws + 256 + 73728;
  // phase 1: cat [B,192,L]  (3,145,728 floats)  -- aliased by branch scratch later
  float* cat  = arena;
  // phase 2 (per branch, sequential reuse):
  float* xz   = arena;                    // [B,256,L]   4,194,304
  float* xc   = xz   + 4194304;           // [B,128,L]   2,097,152
  float* xdbl = xc   + 2097152;           // [B,4,36,L]  2,359,296
  float* ysb  = xdbl + 2359296;           // [B,4,128,L] 8,388,608
  float* yg   = ysb  + 8388608;           // [B,L,128]   2,097,152

  const long long Lc = L4;

  // --- prep + concat + weight repack ---
  pack_cat_k<<<(2*NB*64*L4 + 255)/256, 256, 0, stream>>>(rgb, tin, cat);
  repack_c1w_k<<<(64*1152 + 255)/256, 256, 0, stream>>>(c1w, c1wp);
  prep_bn_k<<<1, 128, 0, stream>>>(c1b, bn1g, bn1b, bn1m, bn1v,
                                   c2b, bn2g, bn2b, bn2m, bn2v,
                                   sc1, sh1, sc2, sh2);

  // --- conv1 3x3 (128->64) implicit-GEMM WMMA (tap-major K), BN+ReLU -> cat[:,0:64,:]
  {
    dim3 g(L4/128, 64/16, NB);
    wmma_gemm_k<true><<<g, 256, 0, stream>>>(
        cat + 64*Lc, c1wp, cat,
        L4, 64, 1152,
        0, 0, 0, 192*Lc,
        1152, 0,
        1, L4, 192*Lc,
        sc1, sh1, nullptr, 0, 1);
  }
  // --- conv2 1x1 (192->64) WMMA, BN+ReLU -> d_out[0]
  {
    dim3 g(L4/128, 64/16, NB);
    wmma_gemm_k<false><<<g, 256, 0, stream>>>(
        cat, c2w, out,
        L4, 64, 192,
        0, 1, L4, 192*Lc,
        192, 0,
        1, L4, 64*Lc,
        sc2, sh2, nullptr, 0, 1);
  }

  // --- two SS2D branches, sequential (reuse arena; residual reads d_in) ---
  for (int br = 0; br < 2; ++br) {
    const int base = 14 + br*11;
    const float* in_w  = (const float*)d_in[base + 0];
    const float* dw_w  = (const float*)d_in[base + 1];
    const float* dw_b  = (const float*)d_in[base + 2];
    const float* xpw   = (const float*)d_in[base + 3];
    const float* dt_w  = (const float*)d_in[base + 4];
    const float* dt_b  = (const float*)d_in[base + 5];
    const float* A_log = (const float*)d_in[base + 6];
    const float* Dsp   = (const float*)d_in[base + 7];
    const float* ln_g  = (const float*)d_in[base + 8];
    const float* ln_b  = (const float*)d_in[base + 9];
    const float* out_w = (const float*)d_in[base + 10];
    const float* X     = br == 0 ? rgb : tin;
    float* outBr       = out + (long long)(1 + br) * NB*64*L4;

    // in_proj: xz[b,n,l] = sum_c X[b,c,l] * in_w[n,c]   (M=L, N=256, K=64)
    {
      dim3 g(L4/128, 256/16, NB);
      wmma_gemm_k<false><<<g, 256, 0, stream>>>(
          X, in_w, xz,
          L4, 256, 64,
          0, 1, L4, 64*Lc,
          64, 0,
          1, L4, 256*Lc,
          nullptr, nullptr, nullptr, 0, 0);
    }
    // depthwise conv + SiLU
    dwconv_silu_k<<<(NB*128*L4 + 255)/256, 256, 0, stream>>>(xz, dw_w, dw_b, xc, NB*128*L4);

    // x_proj, 4 scan directions (row-permutation folded into A addressing)
    for (int kd = 0; kd < 4; ++kd) {
      dim3 g(L4/128, (36 + 15)/16, NB);
      wmma_gemm_k<false><<<g, 256, 0, stream>>>(
          xc, xpw + (long long)kd*36*128, xdbl + (long long)kd*36*L4,
          L4, 36, 128,
          kd, 1, L4, 128*Lc,
          128, 0,
          1, L4, 4*36*Lc,
          nullptr, nullptr, nullptr, 0, 0);
    }

    // selective scan
    scan_k<<<(NB*4*128*16)/256, 256, 0, stream>>>(xc, xdbl, ysb, A_log, dt_w, dt_b, Dsp);

    // combine directions + LayerNorm + SiLU gate -> yg [B,L,128]
    combine_ln_gate_k<<<(NB*L4)/8, 256, 0, stream>>>(ysb, xz, ln_g, ln_b, yg);

    // out_proj + residual -> d_out branch region ([B,64,H,W])
    {
      dim3 g(L4/128, 64/16, NB);
      wmma_gemm_k<false><<<g, 256, 0, stream>>>(
          yg, out_w, outBr,
          L4, 64, 128,
          0, 128, 1, Lc*128,
          128, 0,
          1, L4, 64*Lc,
          nullptr, nullptr, X, 64*Lc, 2);
    }
  }
}
